// SE3TransformerLayer_60730837566348
// MI455X (gfx1250) — compile-verified, matrix operand order checked by reading, not run on hardware
//
#include <hip/hip_runtime.h>
#include <math.h>

// ---------------------------------------------------------------------------
// SE3-transformer-ish attention layer for MI455X (gfx1250, wave32, WMMA).
// All large GEMMs run on v_wmma_f32_16x16x32_bf16 (f32 accumulate).
// Flash-attention over keys so [B,H,N,N] logits are never materialized.
// K tiles are double-buffered into LDS with async global->LDS copies
// (ASYNCcnt) when the toolchain exposes the gfx1250 builtins.
// ---------------------------------------------------------------------------

typedef __bf16 bhalf;
typedef __attribute__((ext_vector_type(16))) __bf16 v16bf;
typedef __attribute__((ext_vector_type(8)))  __bf16 v8bf;
typedef __attribute__((ext_vector_type(8)))  float  v8f;
typedef int v4i __attribute__((vector_size(16)));   // b128 payload type

#define Bd   2
#define Nd   2048
#define Cd   512
#define Hd   8
#define Dh   64
#define NRAD 32

#if defined(__has_builtin)
#if __has_builtin(__builtin_amdgcn_global_load_async_to_lds_b128) && \
    __has_builtin(__builtin_amdgcn_s_wait_asynccnt)
#define USE_ASYNC_LDS 1
#endif
#endif
#ifndef USE_ASYNC_LDS
#define USE_ASYNC_LDS 0
#endif

__device__ __forceinline__ v8f vzero8() {
  v8f z;
#pragma unroll
  for (int i = 0; i < 8; ++i) z[i] = 0.0f;
  return z;
}

__device__ __forceinline__ v8f wmma_bf16(v16bf a, v16bf b, v8f c) {
  // D = A(16x32 bf16) * B(32x16 bf16) + C(16x16 f32)
  return __builtin_amdgcn_wmma_f32_16x16x32_bf16(false, a, false, b,
                                                 (short)0, c, false, false);
}

// A/B fragment load, row-major bf16 source with leading dim `ld` (elements).
// ISA 16-bit A layout: lanes 0-15 -> M=lane, K = 0..7 & 16..23;
// lanes 16-31 -> M=lane-16, K = 8..15 & 24..31.
__device__ __forceinline__ v16bf frag_ld_bf16(const bhalf* p, int ld) {
  const int lane = threadIdx.x & 31;
  const int row  = lane & 15;
  const int kb   = (lane >> 4) << 3;
  const bhalf* r = p + (size_t)row * ld + kb;
  v8bf lo = *(const v8bf*)(r);
  v8bf hi = *(const v8bf*)(r + 16);
  v16bf f;
#pragma unroll
  for (int i = 0; i < 8; ++i) { f[i] = lo[i]; f[i + 8] = hi[i]; }
  return f;
}

// Same fragment, but converting from an f32 row-major source on the fly.
__device__ __forceinline__ v16bf frag_ld_f32(const float* p, int ld) {
  const int lane = threadIdx.x & 31;
  const int row  = lane & 15;
  const int kb   = (lane >> 4) << 3;
  const float* r = p + (size_t)row * ld + kb;
  float4 a0 = *(const float4*)(r);
  float4 a1 = *(const float4*)(r + 4);
  float4 b0 = *(const float4*)(r + 16);
  float4 b1 = *(const float4*)(r + 20);
  v16bf f;
  f[0]  = (bhalf)a0.x; f[1]  = (bhalf)a0.y; f[2]  = (bhalf)a0.z; f[3]  = (bhalf)a0.w;
  f[4]  = (bhalf)a1.x; f[5]  = (bhalf)a1.y; f[6]  = (bhalf)a1.z; f[7]  = (bhalf)a1.w;
  f[8]  = (bhalf)b0.x; f[9]  = (bhalf)b0.y; f[10] = (bhalf)b0.z; f[11] = (bhalf)b0.w;
  f[12] = (bhalf)b1.x; f[13] = (bhalf)b1.y; f[14] = (bhalf)b1.z; f[15] = (bhalf)b1.w;
  return f;
}

#if USE_ASYNC_LDS
// Stage one 32-key x 64-d bf16 K tile (4 KB) global -> LDS asynchronously.
// Each lane copies its key's 128B row as 8x b128 transfers (8 ASYNCcnt incs).
__device__ __forceinline__ void stage_ktile(const bhalf* gK, bhalf* lK) {
  const int lane = threadIdx.x & 31;
  const bhalf* g = gK + (size_t)lane * Cd;   // key = lane, 64 bf16 contiguous
  bhalf* l = lK + lane * Dh;
#pragma unroll
  for (int j = 0; j < 8; ++j) {
    __builtin_amdgcn_global_load_async_to_lds_b128(
        (__attribute__((address_space(1))) v4i*)(g + j * 8),
        (__attribute__((address_space(3))) v4i*)(l + j * 8), 0, 0);
  }
}
#endif

// ---------------------------------------------------------------------------
// Kernel 0: weight prep. W [in][out] f32  ->  WT [out][in] bf16
// (so WMMA B fragments load contiguously in K just like A fragments)
// ---------------------------------------------------------------------------
__global__ __launch_bounds__(256) void wtrans_kernel(const float* __restrict__ W,
                                                     bhalf* __restrict__ WT) {
  int idx = blockIdx.x * 256 + threadIdx.x;     // 512*512 total
  int o = idx >> 9;
  int i = idx & 511;
  WT[(size_t)o * Cd + i] = (bhalf)W[(size_t)i * Cd + o];
}

// ---------------------------------------------------------------------------
// Kernel 1: QKV projection. 16-row tile per WG, 8 waves x 64 cols each.
// which(blockIdx.y): 0 -> Q (pre-scaled by Dh^-1/2), 1 -> K, 2 -> V (stored
// transposed [b][h][d][n] for the PV WMMA).
// ---------------------------------------------------------------------------
__global__ __launch_bounds__(256)
void qkv_kernel(const float* __restrict__ h,
                const bhalf* __restrict__ WqT, const bhalf* __restrict__ WkT,
                const bhalf* __restrict__ WvT,
                const float* __restrict__ bq, const float* __restrict__ bk,
                const float* __restrict__ bv,
                bhalf* __restrict__ Qb, bhalf* __restrict__ Kb,
                bhalf* __restrict__ Vt) {
  const int wave = threadIdx.x >> 5;
  const int lane = threadIdx.x & 31;
  const int row0 = blockIdx.x * 16;          // token-row tile within [0, B*N)
  const int which = blockIdx.y;              // 0=q 1=k 2=v
  const bhalf* WT  = (which == 0) ? WqT : (which == 1) ? WkT : WvT;
  const float* bia = (which == 0) ? bq  : (which == 1) ? bk  : bv;

  v8f acc[4];
#pragma unroll
  for (int j = 0; j < 4; ++j) acc[j] = vzero8();

  for (int kk = 0; kk < Cd / 32; ++kk) {
    v16bf af = frag_ld_f32(h + (size_t)row0 * Cd + kk * 32, Cd);
#pragma unroll
    for (int j = 0; j < 4; ++j) {
      v16bf bf = frag_ld_bf16(WT + (size_t)(wave * 64 + j * 16) * Cd + kk * 32, Cd);
      acc[j] = wmma_bf16(af, bf, acc[j]);
    }
  }

  const int hi = lane >> 4, lo = lane & 15;
  const float scale = (which == 0) ? 0.125f : 1.0f;   // Dh^-0.5 folded into Q
#pragma unroll
  for (int j = 0; j < 4; ++j) {
    int col = wave * 64 + j * 16 + lo;
    float bb = bia[col];
#pragma unroll
    for (int r = 0; r < 8; ++r) {
      int row = row0 + r + 8 * hi;                    // 0..B*N-1
      float val = (acc[j][r] + bb) * scale;
      if (which == 2) {
        int b = row >> 11, n = row & (Nd - 1);
        int hh = col >> 6, d = col & 63;
        Vt[(((size_t)b * Hd + hh) * Dh + d) * Nd + n] = (bhalf)val;
      } else {
        bhalf* dst = (which == 0) ? Qb : Kb;
        dst[(size_t)row * Cd + col] = (bhalf)val;
      }
    }
  }
}

// ---------------------------------------------------------------------------
// Kernel 2: flash attention with radial-MLP logit bias.
// grid = (N/16 query tiles, B); block = 256 (8 waves, wave w == head w).
// Distance + SiLU radial features computed ONCE per (q,k) pair and shared
// across all 8 heads through LDS. K tiles double-buffered via async copies.
// ---------------------------------------------------------------------------
__global__ __launch_bounds__(256)
void attn_kernel(const float* __restrict__ x,
                 const bhalf* __restrict__ Qb, const bhalf* __restrict__ Kb,
                 const bhalf* __restrict__ Vt,
                 const float* __restrict__ Wr1, const float* __restrict__ br1,
                 const float* __restrict__ Wr2, const float* __restrict__ br2,
                 float* __restrict__ attnout) {
  __shared__ float2 s_wr1br1[NRAD];
  __shared__ float4 s_wr2[NRAD][2];
  __shared__ float  s_br2[Hd];
  __shared__ float  s_xq[16][4];
  __shared__ float  s_xk[32][4];
  __shared__ float  s_rw[Hd][16][32];     // radial bias tile, all heads
  __shared__ bhalf  s_p[Hd][16][32];      // P tile (C-layout -> A-layout relay)
#if USE_ASYNC_LDS
  __shared__ bhalf  s_kt[2][Hd][32][Dh];  // double-buffered K tiles (64 KB)
#endif

  const int b    = blockIdx.y;
  const int q0   = blockIdx.x * 16;
  const int wv   = threadIdx.x >> 5;      // head index
  const int lane = threadIdx.x & 31;
  const int hi   = lane >> 4, lo = lane & 15;

  if (threadIdx.x < NRAD) {
    int i = threadIdx.x;
    s_wr1br1[i] = make_float2(Wr1[i], br1[i]);
    s_wr2[i][0] = make_float4(Wr2[i * Hd + 0], Wr2[i * Hd + 1],
                              Wr2[i * Hd + 2], Wr2[i * Hd + 3]);
    s_wr2[i][1] = make_float4(Wr2[i * Hd + 4], Wr2[i * Hd + 5],
                              Wr2[i * Hd + 6], Wr2[i * Hd + 7]);
  }
  if (threadIdx.x < Hd) s_br2[threadIdx.x] = br2[threadIdx.x];
  if (threadIdx.x < 16) {
    const float* xp = x + ((size_t)b * Nd + q0 + threadIdx.x) * 3;
    s_xq[threadIdx.x][0] = xp[0];
    s_xq[threadIdx.x][1] = xp[1];
    s_xq[threadIdx.x][2] = xp[2];
  }

  // Q fragments for this head's 16 queries (loaded once).
  const bhalf* qbase = Qb + ((size_t)(b * Nd + q0) * Cd + wv * Dh);
  v16bf qa0 = frag_ld_bf16(qbase, Cd);
  v16bf qa1 = frag_ld_bf16(qbase + 32, Cd);

  const bhalf* Khead = Kb + ((size_t)b * Nd) * Cd + wv * Dh;
#if USE_ASYNC_LDS
  stage_ktile(Khead, &s_kt[0][wv][0][0]);          // prime first K tile
#endif

  float m[8], l[8];
  v8f O[4];
#pragma unroll
  for (int r = 0; r < 8; ++r) { m[r] = -INFINITY; l[r] = 0.0f; }
#pragma unroll
  for (int c = 0; c < 4; ++c) O[c] = vzero8();

  for (int k0 = 0; k0 < Nd; k0 += 32) {
    __syncthreads();                       // protect s_rw / s_xk reuse
    if (threadIdx.x < 32) {
      const float* xp = x + ((size_t)b * Nd + k0 + threadIdx.x) * 3;
      s_xk[threadIdx.x][0] = xp[0];
      s_xk[threadIdx.x][1] = xp[1];
      s_xk[threadIdx.x][2] = xp[2];
    }
    __syncthreads();

#if USE_ASYNC_LDS
    const int buf = (k0 >> 5) & 1;
    const bool more = (k0 + 32) < Nd;
    if (more)  // kick off next tile; overlaps the radial-MLP VALU phase below
      stage_ktile(Khead + (size_t)(k0 + 32) * Cd, &s_kt[buf ^ 1][wv][0][0]);
#endif

    // ---- radial MLP bias: 512 pairs, 2 per thread, shared across heads ----
#pragma unroll
    for (int pp = 0; pp < 2; ++pp) {
      int pidx = threadIdx.x * 2 + pp;
      int qq = pidx >> 5, kc = pidx & 31;
      float dx = s_xq[qq][0] - s_xk[kc][0];
      float dy = s_xq[qq][1] - s_xk[kc][1];
      float dz = s_xq[qq][2] - s_xk[kc][2];
      float sq = dx * dx + dy * dy + dz * dz;
      float dist = (sq > 0.0f) ? sqrtf(sq) : 0.0f;
      float rw0 = 0.f, rw1 = 0.f, rw2 = 0.f, rw3 = 0.f;
      float rw4 = 0.f, rw5 = 0.f, rw6 = 0.f, rw7 = 0.f;
#pragma unroll 4
      for (int i = 0; i < NRAD; ++i) {
        float2 wb = s_wr1br1[i];
        float a = dist * wb.x + wb.y;
        float t = a / (1.0f + __expf(-a));           // SiLU
        float4 w0 = s_wr2[i][0], w1 = s_wr2[i][1];
        rw0 += t * w0.x; rw1 += t * w0.y; rw2 += t * w0.z; rw3 += t * w0.w;
        rw4 += t * w1.x; rw5 += t * w1.y; rw6 += t * w1.z; rw7 += t * w1.w;
      }
      s_rw[0][qq][kc] = rw0 + s_br2[0];
      s_rw[1][qq][kc] = rw1 + s_br2[1];
      s_rw[2][qq][kc] = rw2 + s_br2[2];
      s_rw[3][qq][kc] = rw3 + s_br2[3];
      s_rw[4][qq][kc] = rw4 + s_br2[4];
      s_rw[5][qq][kc] = rw5 + s_br2[5];
      s_rw[6][qq][kc] = rw6 + s_br2[6];
      s_rw[7][qq][kc] = rw7 + s_br2[7];
    }
    __syncthreads();

    // ---- S = Q K^T for 16q x 32k (this head) ----
    v8f s0 = vzero8(), s1 = vzero8();
#if USE_ASYNC_LDS
    if (more) __builtin_amdgcn_s_wait_asynccnt(8);   // current tile landed
    else      __builtin_amdgcn_s_wait_asynccnt(0);
    const bhalf* kt = &s_kt[buf][wv][0][0];
    s0 = wmma_bf16(qa0, frag_ld_bf16(kt, Dh), s0);
    s0 = wmma_bf16(qa1, frag_ld_bf16(kt + 32, Dh), s0);
    s1 = wmma_bf16(qa0, frag_ld_bf16(kt + 16 * Dh, Dh), s1);
    s1 = wmma_bf16(qa1, frag_ld_bf16(kt + 16 * Dh + 32, Dh), s1);
#else
    const bhalf* kb0 = Khead + (size_t)k0 * Cd;
    const bhalf* kb1 = kb0 + (size_t)16 * Cd;
    __builtin_prefetch(kb0 + (size_t)32 * Cd, 0, 3);   // next key tile
    s0 = wmma_bf16(qa0, frag_ld_bf16(kb0, Cd), s0);
    s0 = wmma_bf16(qa1, frag_ld_bf16(kb0 + 32, Cd), s0);
    s1 = wmma_bf16(qa0, frag_ld_bf16(kb1, Cd), s1);
    s1 = wmma_bf16(qa1, frag_ld_bf16(kb1 + 32, Cd), s1);
#endif

    // ---- add radial bias + online softmax (rows live in 16-lane groups) ----
#pragma unroll
    for (int r = 0; r < 8; ++r) {
      int row = r + 8 * hi;
      float v0 = s0[r] + s_rw[wv][row][lo];
      float v1 = s1[r] + s_rw[wv][row][16 + lo];
      float mx = fmaxf(v0, v1);
#pragma unroll
      for (int msk = 8; msk >= 1; msk >>= 1)
        mx = fmaxf(mx, __shfl_xor(mx, msk, 32));
      float nm = fmaxf(m[r], mx);
      float sc = __expf(m[r] - nm);
      float p0 = __expf(v0 - nm);
      float p1 = __expf(v1 - nm);
      float ps = p0 + p1;
#pragma unroll
      for (int msk = 8; msk >= 1; msk >>= 1)
        ps += __shfl_xor(ps, msk, 32);
      l[r] = l[r] * sc + ps;
      m[r] = nm;
#pragma unroll
      for (int c = 0; c < 4; ++c) O[c][r] *= sc;
      s_p[wv][row][lo]      = (bhalf)p0;
      s_p[wv][row][16 + lo] = (bhalf)p1;
    }

    // ---- O += P V  (A = P 16x32 via LDS relay, B = Vt contiguous in keys) --
    v16bf pa = frag_ld_bf16(&s_p[wv][0][0], 32);
    const bhalf* vb = Vt + ((size_t)b * Hd + wv) * Dh * Nd + k0;
#pragma unroll
    for (int c = 0; c < 4; ++c) {
      v16bf bfrag = frag_ld_bf16(vb + (size_t)(c * 16) * Nd, Nd);
      O[c] = wmma_bf16(pa, bfrag, O[c]);
    }
  }

  // ---- epilogue: O / l -> attnout [b][n][h*64+d] f32 ----
#pragma unroll
  for (int r = 0; r < 8; ++r) {
    float inv = 1.0f / l[r];
    size_t base = ((size_t)b * Nd + q0 + r + 8 * hi) * Cd + wv * Dh + lo;
    attnout[base + 0]  = O[0][r] * inv;
    attnout[base + 16] = O[1][r] * inv;
    attnout[base + 32] = O[2][r] * inv;
    attnout[base + 48] = O[3][r] * inv;
  }
}

// ---------------------------------------------------------------------------
// Kernel 3: out = LayerNorm(h + attnout @ Wo + bo) * gamma + beta.
// 16-row tile per WG held in LDS so LN fuses with the GEMM.
// ---------------------------------------------------------------------------
__global__ __launch_bounds__(256)
void projln_kernel(const float* __restrict__ attnout,
                   const bhalf* __restrict__ WoT, const float* __restrict__ bo,
                   const float* __restrict__ hin,
                   const float* __restrict__ gamma, const float* __restrict__ beta,
                   float* __restrict__ out) {
  __shared__ float ybuf[16][Cd];
  const int wave = threadIdx.x >> 5;
  const int lane = threadIdx.x & 31;
  const int row0 = blockIdx.x * 16;

  v8f acc[4];
#pragma unroll
  for (int j = 0; j < 4; ++j) acc[j] = vzero8();

  for (int kk = 0; kk < Cd / 32; ++kk) {
    v16bf af = frag_ld_f32(attnout + (size_t)row0 * Cd + kk * 32, Cd);
#pragma unroll
    for (int j = 0; j < 4; ++j) {
      v16bf bf = frag_ld_bf16(WoT + (size_t)(wave * 64 + j * 16) * Cd + kk * 32, Cd);
      acc[j] = wmma_bf16(af, bf, acc[j]);
    }
  }

  const int hi = lane >> 4, lo = lane & 15;
#pragma unroll
  for (int j = 0; j < 4; ++j) {
    int col = wave * 64 + j * 16 + lo;
    float bb = bo[col];
#pragma unroll
    for (int r = 0; r < 8; ++r) {
      int rr = r + 8 * hi;
      ybuf[rr][col] = acc[j][r] + bb + hin[(size_t)(row0 + rr) * Cd + col];
    }
  }
  __syncthreads();

  // LayerNorm: 16 threads per row (a row spans one 16-lane shfl group).
  const int r   = threadIdx.x >> 4;
  const int sub = threadIdx.x & 15;
  float sum = 0.0f, ss = 0.0f;
#pragma unroll
  for (int j = 0; j < Cd / 16; ++j) {
    float v = ybuf[r][sub + 16 * j];
    sum += v; ss += v * v;
  }
#pragma unroll
  for (int msk = 8; msk >= 1; msk >>= 1) {
    sum += __shfl_xor(sum, msk, 32);
    ss  += __shfl_xor(ss,  msk, 32);
  }
  const float mean = sum * (1.0f / Cd);
  const float var  = ss * (1.0f / Cd) - mean * mean;
  const float rstd = rsqrtf(var + 1e-5f);
#pragma unroll
  for (int j = 0; j < Cd / 16; ++j) {
    int col = sub + 16 * j;
    float v = ybuf[r][col];
    out[(size_t)(row0 + r) * Cd + col] = (v - mean) * rstd * gamma[col] + beta[col];
  }
}

// ---------------------------------------------------------------------------
extern "C" void kernel_launch(void* const* d_in, const int* in_sizes, int n_in,
                              void* d_out, int out_size, void* d_ws, size_t ws_size,
                              hipStream_t stream) {
  (void)in_sizes; (void)n_in; (void)out_size; (void)ws_size;
  const float* h    = (const float*)d_in[0];
  const float* x    = (const float*)d_in[1];
  /* d_in[2] = pair, unused by the reference */
  const float* Wq   = (const float*)d_in[3];
  const float* bq   = (const float*)d_in[4];
  const float* Wk   = (const float*)d_in[5];
  const float* bk   = (const float*)d_in[6];
  const float* Wv   = (const float*)d_in[7];
  const float* bv   = (const float*)d_in[8];
  const float* Wr1  = (const float*)d_in[9];
  const float* br1  = (const float*)d_in[10];
  const float* Wr2  = (const float*)d_in[11];
  const float* br2  = (const float*)d_in[12];
  const float* Wo   = (const float*)d_in[13];
  const float* bo   = (const float*)d_in[14];
  const float* gam  = (const float*)d_in[15];
  const float* bet  = (const float*)d_in[16];
  float* out = (float*)d_out;

  // Workspace layout (~22 MB total)
  char* ws = (char*)d_ws;
  const size_t wbytes = (size_t)Cd * Cd * sizeof(bhalf);      // 512 KB each
  bhalf* WqT = (bhalf*)(ws);            ws += wbytes;
  bhalf* WkT = (bhalf*)(ws);            ws += wbytes;
  bhalf* WvT = (bhalf*)(ws);            ws += wbytes;
  bhalf* WoT = (bhalf*)(ws);            ws += wbytes;
  const size_t qkv_bytes = (size_t)Bd * Nd * Cd * sizeof(bhalf);  // 4 MB each
  bhalf* Qb = (bhalf*)(ws);             ws += qkv_bytes;
  bhalf* Kb = (bhalf*)(ws);             ws += qkv_bytes;
  bhalf* Vt = (bhalf*)(ws);             ws += qkv_bytes;          // [b][h][d][n]
  float* attnout = (float*)(ws);        ws += (size_t)Bd * Nd * Cd * sizeof(float);

  // 0) weight prep (bf16 transposes)
  wtrans_kernel<<<(Cd * Cd) / 256, 256, 0, stream>>>(Wq, WqT);
  wtrans_kernel<<<(Cd * Cd) / 256, 256, 0, stream>>>(Wk, WkT);
  wtrans_kernel<<<(Cd * Cd) / 256, 256, 0, stream>>>(Wv, WvT);
  wtrans_kernel<<<(Cd * Cd) / 256, 256, 0, stream>>>(Wo, WoT);

  // 1) QKV projection
  qkv_kernel<<<dim3((Bd * Nd) / 16, 3), 256, 0, stream>>>(
      h, WqT, WkT, WvT, bq, bk, bv, Qb, Kb, Vt);

  // 2) flash attention + radial bias
  attn_kernel<<<dim3(Nd / 16, Bd), 256, 0, stream>>>(
      x, Qb, Kb, Vt, Wr1, br1, Wr2, br2, attnout);

  // 3) output projection + residual + LayerNorm
  projln_kernel<<<(Bd * Nd) / 16, 256, 0, stream>>>(
      attnout, WoT, bo, h, gam, bet, out);
}
